// NeighborConsensusEncoder_87239375716499
// MI455X (gfx1250) — compile-verified, matrix operand order checked by reading, not run on hardware
//
#include <hip/hip_runtime.h>

typedef __attribute__((ext_vector_type(16))) _Float16 v16h;
typedef __attribute__((ext_vector_type(8)))  _Float16 v8h;
typedef __attribute__((ext_vector_type(8)))  float    v8f;
typedef __attribute__((ext_vector_type(4)))  float    v4f;

#define NB   8
#define NC   10
#define NH   384
#define NW   384
#define HW   (NH*NW)        // 147456
#define ND   64
#define LOGC 2.302585092994046f

// ---------------------------------------------------------------------------
// K0: zero small stats scratch
// ---------------------------------------------------------------------------
__global__ void k_zero(float* p, int n) {
    int i = blockIdx.x * blockDim.x + threadIdx.x;
    if (i < n) p[i] = 0.0f;
}

// ---------------------------------------------------------------------------
// K0b: pre-swizzle W2 (f32, 64x64 row-major [out_d][in_c]) into lane-ordered
// f16 WMMA B fragments (8 frags x 32 lanes x 16 f16 = 8 KB, L2-resident).
// ---------------------------------------------------------------------------
__global__ void k_prep_w2(const float* __restrict__ W2, _Float16* __restrict__ wfrag) {
    int i = threadIdx.x;              // 256 threads = 8 frags x 32 lanes
    int fragidx = i >> 5;             // 0..7  (nt*2 + kk)
    int lane    = i & 31;
    int nt = fragidx >> 1, kk = fragidx & 1;
    int m    = lane & 15;
    int koff = (lane < 16) ? 0 : 8;
    int nrow = nt * 16 + m;
    _Float16* dst = wfrag + (size_t)i * 16;
    #pragma unroll
    for (int half = 0; half < 2; half++)
        #pragma unroll
        for (int j = 0; j < 8; j++)
            dst[half * 8 + j] = (_Float16)W2[nrow * ND + kk * 32 + koff + half * 16 + j];
}

// ---------------------------------------------------------------------------
// K1: softmax + neighbor features + 1x1 conv (8->64) + GN partial stats
// block = 256 threads = one 16x16 pixel tile of one batch image
// ---------------------------------------------------------------------------
__global__ __launch_bounds__(256) void k_feat_fc1(
    const float* __restrict__ x,      // (B, 10, H, W)
    const float* __restrict__ W1,     // (64, 8)
    const float* __restrict__ b1,     // (64)
    _Float16*    __restrict__ h16,    // (B*HW, 64)
    float*       __restrict__ stats)  // (B, 8, 2)  {sum, sumsq}
{
    __shared__ float sP[400][NC];     // 20x20 softmax tile, 16 KB
    __shared__ float sW1[ND * 8];
    __shared__ float sb1[ND];
    __shared__ float sSum[8], sSq[8];

    const int tid = threadIdx.x;
    const int tilesPerImg = (NH/16) * (NW/16);     // 576
    const int b  = blockIdx.x / tilesPerImg;
    const int t  = blockIdx.x % tilesPerImg;
    const int ty0 = (t / (NW/16)) * 16;
    const int tx0 = (t % (NW/16)) * 16;

    for (int i = tid; i < ND * 8; i += 256) sW1[i] = W1[i];
    if (tid < ND) sb1[tid] = b1[tid];
    if (tid < 8) { sSum[tid] = 0.0f; sSq[tid] = 0.0f; }

    // -------- softmax into LDS (20x20 halo tile, replicate padding) --------
    const float* xb = x + (size_t)b * NC * HW;
    for (int e = tid; e < 400; e += 256) {
        int ly = e / 20, lx = e % 20;
        int gy = ty0 + ly - 2; gy = gy < 0 ? 0 : (gy > NH-1 ? NH-1 : gy);
        int gx = tx0 + lx - 2; gx = gx < 0 ? 0 : (gx > NW-1 ? NW-1 : gx);
        size_t off = (size_t)gy * NW + gx;
        float v[NC];
        float mx = -1e30f;
        #pragma unroll
        for (int c = 0; c < NC; c++) { v[c] = xb[(size_t)c * HW + off]; mx = fmaxf(mx, v[c]); }
        float s = 0.0f;
        #pragma unroll
        for (int c = 0; c < NC; c++) { v[c] = __expf(v[c] - mx); s += v[c]; }
        float inv = 1.0f / s;
        #pragma unroll
        for (int c = 0; c < NC; c++) sP[e][c] = v[c] * inv;
    }
    __syncthreads();

    // -------- per-pixel window sums (5x5 outer, 3x3 inner) --------
    const int ly = tid / 16, lx = tid % 16;
    float S5[NC], Q5[NC], S3[NC], Q3[NC];
    #pragma unroll
    for (int c = 0; c < NC; c++) { S5[c] = Q5[c] = S3[c] = Q3[c] = 0.0f; }

    for (int dy = 0; dy < 5; dy++) {
        for (int dx = 0; dx < 5; dx++) {
            const float* p = sP[(ly + dy) * 20 + (lx + dx)];
            const bool inner = (dy >= 1 && dy <= 3 && dx >= 1 && dx <= 3);
            #pragma unroll
            for (int c = 0; c < NC; c++) {
                float pv = p[c];
                float pq = pv * pv;
                S5[c] += pv; Q5[c] += pq;
                if (inner) { S3[c] += pv; Q3[c] += pq; }
            }
        }
    }
    float pc[NC];
    {
        const float* pctr = sP[(ly + 2) * 20 + (lx + 2)];
        #pragma unroll
        for (int c = 0; c < NC; c++) pc[c] = pctr[c];
    }

    // -------- features: [match, maj, uni, var] for k=3 then k=5 --------
    float f[8];
    {
        const float* Ss[2] = { S3, S5 };
        const float* Qs[2] = { Q3, Q5 };
        const float  Kk[2] = { 9.0f, 25.0f };
        #pragma unroll
        for (int s = 0; s < 2; s++) {
            const float* S = Ss[s];
            const float* Q = Qs[s];
            const float K = Kk[s], Kinv = 1.0f / K;
            float match = 0.0f, mx = -1e30f, ent = 0.0f, var = 0.0f;
            #pragma unroll
            for (int c = 0; c < NC; c++) {
                match += pc[c] * (S[c] - pc[c]);
                mx = fmaxf(mx, S[c]);
                float pb  = S[c] * Kinv;
                float pcl = fmaxf(pb, 1e-8f);
                ent -= pcl * __logf(pcl);
                var += Q[c] * Kinv - pb * pb;
            }
            f[4*s + 0] = match / (K - 1.0f);
            f[4*s + 1] = mx * Kinv;
            f[4*s + 2] = 1.0f - ent * (1.0f / LOGC);
            f[4*s + 3] = var;
        }
    }

    // -------- 1x1 conv 8 -> 64 + partial GN stats, store h as f16 --------
    const int pixIdx = (ty0 + ly) * NW + (tx0 + lx);
    _Float16* hp = h16 + ((size_t)b * HW + pixIdx) * ND;
    float gs[8], gq[8];
    #pragma unroll
    for (int g = 0; g < 8; g++) { gs[g] = 0.0f; gq[g] = 0.0f; }

    #pragma unroll 8
    for (int d = 0; d < ND; d++) {
        float acc = sb1[d];
        #pragma unroll
        for (int c = 0; c < 8; c++) acc += f[c] * sW1[d * 8 + c];
        int g = d >> 3;
        gs[g] += acc;
        gq[g] += acc * acc;
        hp[d] = (_Float16)acc;
    }

    #pragma unroll
    for (int g = 0; g < 8; g++) {
        atomicAdd(&sSum[g], gs[g]);
        atomicAdd(&sSq[g],  gq[g]);
    }
    __syncthreads();
    if (tid < 8) {
        atomicAdd(&stats[((size_t)b * 8 + tid) * 2 + 0], sSum[tid]);
        atomicAdd(&stats[((size_t)b * 8 + tid) * 2 + 1], sSq[tid]);
    }
}

// ---------------------------------------------------------------------------
// K2: fold GN stats into per-(b, channel) scale/bias
// ---------------------------------------------------------------------------
__global__ void k_gnparams(const float* __restrict__ stats,
                           const float* __restrict__ gn_w,
                           const float* __restrict__ gn_b,
                           float* __restrict__ scalebias) // (B, 64, 2)
{
    int i = blockIdx.x * blockDim.x + threadIdx.x;
    if (i >= NB * ND) return;
    int b = i >> 6, k = i & 63, g = k >> 3;
    const float n = 8.0f * (float)HW;
    float sum = stats[((size_t)b * 8 + g) * 2 + 0];
    float sq  = stats[((size_t)b * 8 + g) * 2 + 1];
    float mu  = sum / n;
    float var = sq / n - mu * mu;
    float rs  = rsqrtf(var + 1e-5f);
    float sc  = rs * gn_w[k];
    scalebias[i * 2 + 0] = sc;
    scalebias[i * 2 + 1] = gn_b[k] - mu * sc;
}

// ---------------------------------------------------------------------------
// K3: GN apply + erf-GELU + 1x1 conv 64->64 via v_wmma_f32_16x16x32_f16
// block = 256 threads = 8 waves; each wave owns 16 pixels x all 64 outputs
// ---------------------------------------------------------------------------
__global__ __launch_bounds__(256) void k_gelu_fc2(
    const _Float16* __restrict__ h16,       // (B*HW, 64) f16
    const float*    __restrict__ scalebias, // (B, 64, 2)
    const _Float16* __restrict__ wfrag,     // (8 frags, 32 lanes, 16) f16
    const float*    __restrict__ b2,        // (64)
    float*          __restrict__ out)       // (B, 64, H, W)
{
    // row stride 132 floats: 528B per row => 16B-aligned b128 LDS ops,
    // lane->bank stride of 4 keeps lanes 0-15 fully conflict-free.
    __shared__ float sOut[ND][132];         // 64 x 128 pixels (+4 pad), 33 KB
    __shared__ float sSB[ND * 2];

    const int tid  = threadIdx.x;
    const int wave = tid >> 5;
    const int lane = tid & 31;
    const int blocksPerImg = HW / 128;      // 1152
    const int b        = blockIdx.x / blocksPerImg;
    const int pixBlock = (blockIdx.x % blocksPerImg) * 128;

    if (tid < ND * 2) sSB[tid] = scalebias[(size_t)b * ND * 2 + tid];
    __syncthreads();

    const int m    = lane & 15;                   // pixel row within 16-tile
    const int koff = (lane < 16) ? 0 : 8;         // K sub-offset per A layout
    const int pix  = pixBlock + wave * 16 + m;
    const _Float16* hp = h16 + ((size_t)b * HW + pix) * ND;

    // ---- build A fragments (GN + exact GELU applied in f32, repack f16) ----
    v16h A[2];
    #pragma unroll
    for (int kk = 0; kk < 2; kk++) {
        const int kb = kk * 32;
        #pragma unroll
        for (int half = 0; half < 2; half++) {
            const int k0 = kb + koff + half * 16;
            v8h raw = *(const v8h*)(hp + k0);
            #pragma unroll
            for (int j = 0; j < 8; j++) {
                int ch = k0 + j;
                float v = (float)raw[j];
                v = v * sSB[ch * 2 + 0] + sSB[ch * 2 + 1];
                v = 0.5f * v * (1.0f + erff(v * 0.70710678118654752f));
                A[kk][half * 8 + j] = (_Float16)v;
            }
        }
    }

    // ---- 4 N-tiles x 2 K-steps of WMMA; B fragments pre-swizzled in f16 ----
    const int pl  = wave * 16;
    const int mm0 = (lane < 16) ? 0 : 8;          // this lane's 8 consecutive M
    #pragma unroll
    for (int nt = 0; nt < 4; nt++) {
        v8f c = {};
        #pragma unroll
        for (int kk = 0; kk < 2; kk++) {
            v16h Bf = *(const v16h*)(wfrag + (size_t)((nt * 2 + kk) * 32 + lane) * 16);
            c = __builtin_amdgcn_wmma_f32_16x16x32_f16(
                    false, A[kk], false, Bf, (short)0, c, false, false);
        }
        // C layout: VGPR r -> (M=mm0+r, N=lane&15); 8 consecutive M per lane
        // => two aligned b128 LDS stores instead of 8 scalar stores.
        const int nn = nt * 16 + (lane & 15);
        float* dst = &sOut[nn][pl + mm0];
        *(v4f*)(dst + 0) = (v4f){ c[0], c[1], c[2], c[3] };
        *(v4f*)(dst + 4) = (v4f){ c[4], c[5], c[6], c[7] };
    }
    __syncthreads();

    // ---- coalesced channel-major store (+bias): b128 LDS loads + stores ----
    float* ob = out + (size_t)b * ND * HW;
    for (int i = tid; i < ND * 32; i += 256) {
        int d  = i >> 5;
        int p4 = (i & 31) * 4;
        float bias = b2[d];
        v4f v = *(const v4f*)(&sOut[d][p4]);
        v.x += bias; v.y += bias; v.z += bias; v.w += bias;
        *(v4f*)(ob + (size_t)d * HW + pixBlock + p4) = v;
    }
}

// ---------------------------------------------------------------------------
extern "C" void kernel_launch(void* const* d_in, const int* in_sizes, int n_in,
                              void* d_out, int out_size, void* d_ws, size_t ws_size,
                              hipStream_t stream) {
    const float* x    = (const float*)d_in[0];
    const float* W1   = (const float*)d_in[1];
    const float* b1   = (const float*)d_in[2];
    const float* gn_w = (const float*)d_in[3];
    const float* gn_b = (const float*)d_in[4];
    const float* W2   = (const float*)d_in[5];
    const float* b2   = (const float*)d_in[6];
    float* out = (float*)d_out;

    // workspace layout
    float*    stats     = (float*)d_ws;                        // 128 f32 @ 0
    float*    scalebias = (float*)((char*)d_ws + 512);         // 1024 f32
    _Float16* wfrag     = (_Float16*)((char*)d_ws + 8192);     // 8 KB swizzled W2
    _Float16* h16       = (_Float16*)((char*)d_ws + 16384);    // 151 MB

    k_zero<<<1, 128, 0, stream>>>(stats, 128);
    k_prep_w2<<<1, 256, 0, stream>>>(W2, wfrag);

    const int tilesPerImg = (NH/16) * (NW/16);                 // 576
    k_feat_fc1<<<NB * tilesPerImg, 256, 0, stream>>>(x, W1, b1, h16, stats);

    k_gnparams<<<2, 256, 0, stream>>>(stats, gn_w, gn_b, scalebias);

    const int blocksPerImg = HW / 128;                         // 1152
    k_gelu_fc2<<<NB * blocksPerImg, 256, 0, stream>>>(h16, scalebias, wfrag, b2, out);
}
